// HybridBlock_34754875359676
// MI455X (gfx1250) — matrix operand order repaired and checked by reading, not compiled, not run on hardware
//
#include <hip/hip_runtime.h>
#include <math.h>

// ---- problem dims (fixed by reference) ----
#define Bb 4
#define Tt 1024
#define Dd 1024
#define Hh 8192
#define EPSF 1e-6f
#define PLIF_TH_F 0.05f

typedef unsigned short u16;
typedef __bf16 bf16_t;
typedef bf16_t v16bf __attribute__((ext_vector_type(16)));
typedef float  f32x8 __attribute__((ext_vector_type(8)));
typedef unsigned int u32x8 __attribute__((ext_vector_type(8)));
typedef __attribute__((address_space(3))) u16 lds_u16;

// ---- helpers ----
__device__ __forceinline__ u16 f2bf(float f) {
  union { float f; unsigned u; } x; x.f = f;
  unsigned r = x.u + 0x7FFFu + ((x.u >> 16) & 1u);   // RNE
  return (u16)(r >> 16);
}

__device__ __forceinline__ f32x8 wmma_bf16(v16bf a, v16bf b, f32x8 c) {
  // v_wmma_f32_16x16x32_bf16 : D = A(16x32) x B(32x16) + C
  return __builtin_amdgcn_wmma_f32_16x16x32_bf16(false, a, false, b, (short)0, c, false, false);
}

// A fragment: X row-major (M x K), tile rows [row0,row0+16), K [k0,k0+32)
// Contiguous k order used here (not ISA table order): lanes 0-15 row=lane, elems0-15 = K k0..k0+15 contiguous
//             lanes 16-31 row=lane-16, K k0+16..k0+31.
__device__ __forceinline__ v16bf ldA(const u16* __restrict__ X, int ld, int row0, int k0) {
  int lane = threadIdx.x & 31;
  const u16* p = X + (size_t)(row0 + (lane & 15)) * ld + (k0 + ((lane >> 4) << 4));
  uint4 lo = *(const uint4*)p;
  uint4 hi = *(const uint4*)(p + 8);
  u32x8 u; u[0]=lo.x; u[1]=lo.y; u[2]=lo.z; u[3]=lo.w; u[4]=hi.x; u[5]=hi.y; u[6]=hi.z; u[7]=hi.w;
  return __builtin_bit_cast(v16bf, u);
}

// B fragment for C = X * W^T : B[k][n] = W[n][k], W row-major (N x K).
// Contiguous k order used here, same as ldA (not ISA table order): lanes 0-15 col=lane hold K k0..k0+15 contiguous; lanes 16-31 K k0+16..k0+31.
__device__ __forceinline__ v16bf ldB(const u16* __restrict__ W, int ld, int col0, int k0) {
  int lane = threadIdx.x & 31;
  const u16* p = W + (size_t)(col0 + (lane & 15)) * ld + (k0 + ((lane >> 4) << 4));
  u32x8 u = *(const u32x8*)p;
  return __builtin_bit_cast(v16bf, u);
}

// D store: lanes 0-15 col=lane rows row0..row0+7; lanes 16-31 col=lane-16 rows row0+8..+15
__device__ __forceinline__ void stD(float* __restrict__ C, int ld, int row0, int col0, f32x8 d) {
  int lane = threadIdx.x & 31;
  int nn = col0 + (lane & 15);
  int mb = row0 + ((lane >> 4) << 3);
#pragma unroll
  for (int i = 0; i < 8; ++i) C[(size_t)(mb + i) * ld + nn] = d[i];
}

// ---- prep kernels ----
__global__ void k_f32_to_bf16(const float* __restrict__ in, u16* __restrict__ out, int n) {
  int stride = gridDim.x * blockDim.x;
  for (int i = blockIdx.x * blockDim.x + threadIdx.x; i < n; i += stride) out[i] = f2bf(in[i]);
}

__global__ void k_masked_bf16(const float* __restrict__ w, const unsigned char* __restrict__ m,
                              u16* __restrict__ out, int n) {
  int stride = gridDim.x * blockDim.x;
  for (int i = blockIdx.x * blockDim.x + threadIdx.x; i < n; i += stride)
    out[i] = m[i] ? f2bf(w[i]) : (u16)0;
}

__global__ void k_init(u16* __restrict__ hA, u16* __restrict__ hB, int* __restrict__ bar) {
  int stride = gridDim.x * blockDim.x;
  for (int i = blockIdx.x * blockDim.x + threadIdx.x; i < 16 * Dd; i += stride) { hA[i] = 0; hB[i] = 0; }
  if (blockIdx.x == 0 && threadIdx.x == 0) { bar[0] = 0; bar[1] = 0; }
}

// ---- rmsnorm: one 256-thread block per row, f32 in -> bf16 out ----
__global__ void k_rmsnorm(const float* __restrict__ x, const float* __restrict__ w,
                          u16* __restrict__ out) {
  const float* row = x + (size_t)blockIdx.x * Dd;
  u16* orow = out + (size_t)blockIdx.x * Dd;
  float ss = 0.f;
  for (int i = threadIdx.x; i < Dd; i += blockDim.x) { float v = row[i]; ss += v * v; }
#pragma unroll
  for (int off = 16; off > 0; off >>= 1) ss += __shfl_down(ss, off, 32);
  __shared__ float red[8];
  __shared__ float scale;
  int lane = threadIdx.x & 31, wid = threadIdx.x >> 5;
  if (lane == 0) red[wid] = ss;
  __syncthreads();
  if (threadIdx.x == 0) {
    float t = 0.f;
    for (int i = 0; i < 8; ++i) t += red[i];
    scale = rsqrtf(t / (float)Dd + EPSF);
  }
  __syncthreads();
  float sc = scale;
  for (int i = threadIdx.x; i < Dd; i += blockDim.x) orow[i] = f2bf(row[i] * sc * w[i]);
}

// ---- U = a @ Wi^T  (4096x1024x1024), f32 out. block 256thr, tile 128x128, wave 32x64 ----
__global__ void k_gemm_u(const u16* __restrict__ A, const u16* __restrict__ W, float* __restrict__ C) {
  int wave = threadIdx.x >> 5;
  int row0 = blockIdx.y * 128 + (wave >> 1) * 32;
  int col0 = blockIdx.x * 128 + (wave & 1) * 64;
  f32x8 acc[2][4] = {};
  for (int k0 = 0; k0 < Dd; k0 += 32) {
    v16bf a0 = ldA(A, Dd, row0, k0);
    v16bf a1 = ldA(A, Dd, row0 + 16, k0);
#pragma unroll
    for (int c = 0; c < 4; ++c) {
      v16bf b = ldB(W, Dd, col0 + 16 * c, k0);
      acc[0][c] = wmma_bf16(a0, b, acc[0][c]);
      acc[1][c] = wmma_bf16(a1, b, acc[1][c]);
    }
  }
#pragma unroll
  for (int r = 0; r < 2; ++r)
#pragma unroll
    for (int c = 0; c < 4; ++c) stD(C, Dd, row0 + 16 * r, col0 + 16 * c, acc[r][c]);
}

// ---- persistent liquid+PLIF scan ----
// grid = 64 blocks (16 cols each), blockDim = 64 (2 waves). Wr tile (16x1024 bf16 = 32KB)
// staged into LDS with CDNA5 async global->LDS DMA (ASYNCcnt). h ping-pong bf16 buffers
// 16x1024 (rows 4..15 = 0). One atomic grid barrier per timestep.
__global__ void __launch_bounds__(64) k_liquid_plif(
    const u16* __restrict__ Wr, const float* __restrict__ U,
    const float* __restrict__ liq_b, const float* __restrict__ liq_tau,
    const float* __restrict__ plif_tau,
    u16* __restrict__ hA, u16* __restrict__ hB,
    u16* __restrict__ Sout, int* __restrict__ bar) {
  __shared__ u16 wr_tile[16 * Dd];
  __shared__ float part[32 * 8];
  int tid = threadIdx.x;
  int wave = tid >> 5, lane = tid & 31;
  int col0 = blockIdx.x * 16;

  // stage this block's Wr rows (= output cols) into LDS via async DMA:
  // each lane moves 16B per iteration; tracked by ASYNCcnt, not LOADcnt.
  {
    unsigned lds_base = (unsigned)(size_t)(lds_u16*)wr_tile;
    const u16* gbase = Wr + (size_t)col0 * Dd;
    for (int i = tid; i < 16 * Dd / 8; i += blockDim.x) {
      unsigned lds_addr = lds_base + (unsigned)i * 16u;
      const u16* gp = gbase + (size_t)i * 8;
      asm volatile("global_load_async_to_lds_b128 %0, %1, off"
                   :: "v"(lds_addr), "v"(gp) : "memory");
    }
    asm volatile("s_wait_asynccnt 0" ::: "memory");
  }

  int n = col0 + (lane & 15);
  float dec  = __expf(-1.f / liq_tau[n]);
  float bb   = liq_b[n];
  float itau = 1.f / plif_tau[0];
  float hst[4] = {0.f, 0.f, 0.f, 0.f};
  float vst[4] = {0.f, 0.f, 0.f, 0.f};
  __syncthreads();

  int kbase = wave * (Dd / 2);   // split K reduction across the 2 waves
  for (int t = 0; t < Tt; ++t) {
    const u16* hprev = (t & 1) ? hB : hA;
    u16* hnext = (t & 1) ? hA : hB;
    f32x8 acc = {};
    for (int k0 = 0; k0 < Dd / 2; k0 += 32) {
      v16bf a = ldA(hprev, Dd, 0, kbase + k0);
      const u16* p = &wr_tile[(lane & 15) * Dd + (kbase + k0 + ((lane >> 4) << 4))];
      v16bf b = __builtin_bit_cast(v16bf, *(const u32x8*)p);   // ds_load fed WMMA
      acc = wmma_bf16(a, b, acc);
    }
    if (wave == 1) {
#pragma unroll
      for (int i = 0; i < 8; ++i) part[lane * 8 + i] = acc[i];
    }
    __syncthreads();
    if (wave == 0 && lane < 16) {
#pragma unroll
      for (int i = 0; i < 4; ++i) {        // real batch rows 0..3
        float pre = acc[i] + part[lane * 8 + i] + U[((size_t)i * Tt + t) * Dd + n] + bb;
        float f = tanhf(pre);
        float h2 = hst[i] * dec + (1.f - dec) * f;
        hst[i] = h2;
        hnext[i * Dd + n] = f2bf(h2);
        float v = vst[i] + (h2 - vst[i]) * itau;          // PLIF fused
        float sp = (v - PLIF_TH_F) > 0.f ? 1.f : 0.f;
        vst[i] = v - sp * PLIF_TH_F;
        Sout[((size_t)i * Tt + t) * Dd + n] = f2bf(sp);
      }
    }
    __threadfence();
    __syncthreads();
    if (tid == 0) {                                        // grid barrier
      int g = __hip_atomic_load(&bar[1], __ATOMIC_RELAXED, __HIP_MEMORY_SCOPE_AGENT);
      int a = __hip_atomic_fetch_add(&bar[0], 1, __ATOMIC_ACQ_REL, __HIP_MEMORY_SCOPE_AGENT);
      if (a == (int)gridDim.x - 1) {
        __hip_atomic_store(&bar[0], 0, __ATOMIC_RELAXED, __HIP_MEMORY_SCOPE_AGENT);
        __hip_atomic_fetch_add(&bar[1], 1, __ATOMIC_RELEASE, __HIP_MEMORY_SCOPE_AGENT);
      } else {
        while (__hip_atomic_load(&bar[1], __ATOMIC_ACQUIRE, __HIP_MEMORY_SCOPE_AGENT) == g)
          __builtin_amdgcn_s_sleep(1);
      }
    }
    __syncthreads();
  }
}

// ---- fused syn + gate GEMMs + gated residual: Xmid = X + (S@Wsyn^T)*sigmoid(S@Wgate^T+gb)
// block 256 thr, tile 128x64, wave 32x32 per GEMM (dual accumulators) ----
__global__ void k_syn_gate(const u16* __restrict__ S, const u16* __restrict__ Wsyn,
                           const u16* __restrict__ Wgate, const float* __restrict__ gate_b,
                           const float* __restrict__ X, float* __restrict__ Xmid) {
  int wave = threadIdx.x >> 5, lane = threadIdx.x & 31;
  int row0 = blockIdx.y * 128 + (wave >> 1) * 32;
  int col0 = blockIdx.x * 64 + (wave & 1) * 32;
  f32x8 as[2][2] = {}, ag[2][2] = {};
  for (int k0 = 0; k0 < Dd; k0 += 32) {
    v16bf a0 = ldA(S, Dd, row0, k0);
    v16bf a1 = ldA(S, Dd, row0 + 16, k0);
    v16bf bs0 = ldB(Wsyn, Dd, col0, k0);
    v16bf bs1 = ldB(Wsyn, Dd, col0 + 16, k0);
    v16bf bg0 = ldB(Wgate, Dd, col0, k0);
    v16bf bg1 = ldB(Wgate, Dd, col0 + 16, k0);
    as[0][0] = wmma_bf16(a0, bs0, as[0][0]); as[0][1] = wmma_bf16(a0, bs1, as[0][1]);
    as[1][0] = wmma_bf16(a1, bs0, as[1][0]); as[1][1] = wmma_bf16(a1, bs1, as[1][1]);
    ag[0][0] = wmma_bf16(a0, bg0, ag[0][0]); ag[0][1] = wmma_bf16(a0, bg1, ag[0][1]);
    ag[1][0] = wmma_bf16(a1, bg0, ag[1][0]); ag[1][1] = wmma_bf16(a1, bg1, ag[1][1]);
  }
#pragma unroll
  for (int r = 0; r < 2; ++r)
#pragma unroll
    for (int c = 0; c < 2; ++c) {
      int nn = col0 + 16 * c + (lane & 15);
      int mb = row0 + 16 * r + ((lane >> 4) << 3);
      float gb = gate_b[nn];
#pragma unroll
      for (int i = 0; i < 8; ++i) {
        size_t idx = (size_t)(mb + i) * Dd + nn;
        float gate = 1.f / (1.f + __expf(-(ag[r][c][i] + gb)));
        Xmid[idx] = X[idx] + as[r][c][i] * gate;
      }
    }
}

// ---- fused FFN up: T = silu(Y@Wg^T) * (Y@Wu^T), bf16 out. tile 128x64, wave 32x32 dual ----
__global__ void k_ffn_up(const u16* __restrict__ Y, const u16* __restrict__ Wg,
                         const u16* __restrict__ Wu, u16* __restrict__ Tb) {
  int wave = threadIdx.x >> 5, lane = threadIdx.x & 31;
  int row0 = blockIdx.y * 128 + (wave >> 1) * 32;
  int col0 = blockIdx.x * 64 + (wave & 1) * 32;
  f32x8 agv[2][2] = {}, auv[2][2] = {};
  for (int k0 = 0; k0 < Dd; k0 += 32) {
    v16bf a0 = ldA(Y, Dd, row0, k0);
    v16bf a1 = ldA(Y, Dd, row0 + 16, k0);
    v16bf bg0 = ldB(Wg, Dd, col0, k0);
    v16bf bg1 = ldB(Wg, Dd, col0 + 16, k0);
    v16bf bu0 = ldB(Wu, Dd, col0, k0);
    v16bf bu1 = ldB(Wu, Dd, col0 + 16, k0);
    agv[0][0] = wmma_bf16(a0, bg0, agv[0][0]); agv[0][1] = wmma_bf16(a0, bg1, agv[0][1]);
    agv[1][0] = wmma_bf16(a1, bg0, agv[1][0]); agv[1][1] = wmma_bf16(a1, bg1, agv[1][1]);
    auv[0][0] = wmma_bf16(a0, bu0, auv[0][0]); auv[0][1] = wmma_bf16(a0, bu1, auv[0][1]);
    auv[1][0] = wmma_bf16(a1, bu0, auv[1][0]); auv[1][1] = wmma_bf16(a1, bu1, auv[1][1]);
  }
#pragma unroll
  for (int r = 0; r < 2; ++r)
#pragma unroll
    for (int c = 0; c < 2; ++c) {
      int nn = col0 + 16 * c + (lane & 15);
      int mb = row0 + 16 * r + ((lane >> 4) << 3);
#pragma unroll
      for (int i = 0; i < 8; ++i) {
        float g = agv[r][c][i], u = auv[r][c][i];
        float t = (g / (1.f + __expf(-g))) * u;   // silu(g)*u
        Tb[(size_t)(mb + i) * Hh + nn] = f2bf(t);
      }
    }
}

// ---- FFN down + residual: Out = Xmid + T @ Wd^T. K=8192, tile 128x128, wave 32x64 ----
__global__ void k_ffn_down(const u16* __restrict__ Tb, const u16* __restrict__ Wd,
                           const float* __restrict__ Xmid, float* __restrict__ Out) {
  int wave = threadIdx.x >> 5, lane = threadIdx.x & 31;
  int row0 = blockIdx.y * 128 + (wave >> 1) * 32;
  int col0 = blockIdx.x * 128 + (wave & 1) * 64;
  f32x8 acc[2][4] = {};
  for (int k0 = 0; k0 < Hh; k0 += 32) {
    // pull the deep-K weight stream ahead of the WMMA pipeline (global_prefetch)
    if (k0 + 128 < Hh)
      __builtin_prefetch(Wd + (size_t)(col0 + (lane & 15)) * Hh + k0 + 128, 0, 1);
    v16bf a0 = ldA(Tb, Hh, row0, k0);
    v16bf a1 = ldA(Tb, Hh, row0 + 16, k0);
#pragma unroll
    for (int c = 0; c < 4; ++c) {
      v16bf b = ldB(Wd, Hh, col0 + 16 * c, k0);
      acc[0][c] = wmma_bf16(a0, b, acc[0][c]);
      acc[1][c] = wmma_bf16(a1, b, acc[1][c]);
    }
  }
#pragma unroll
  for (int r = 0; r < 2; ++r)
#pragma unroll
    for (int c = 0; c < 4; ++c) {
      int nn = col0 + 16 * c + (lane & 15);
      int mb = row0 + 16 * r + ((lane >> 4) << 3);
#pragma unroll
      for (int i = 0; i < 8; ++i) {
        size_t idx = (size_t)(mb + i) * Dd + nn;
        Out[idx] = Xmid[idx] + acc[r][c][i];
      }
    }
}

// ---- host launcher ----
extern "C" void kernel_launch(void* const* d_in, const int* in_sizes, int n_in,
                              void* d_out, int out_size, void* d_ws, size_t ws_size,
                              hipStream_t stream) {
  (void)in_sizes; (void)n_in; (void)out_size; (void)ws_size;
  const float* x        = (const float*)d_in[0];
  const float* ln1_w    = (const float*)d_in[1];
  const float* liq_Wi   = (const float*)d_in[2];
  const float* liq_Wr   = (const float*)d_in[3];
  const float* liq_b    = (const float*)d_in[4];
  const float* liq_tau  = (const float*)d_in[5];
  const float* plif_tau = (const float*)d_in[6];
  const float* syn_w    = (const float*)d_in[7];
  const float* gate_w   = (const float*)d_in[8];
  const float* gate_b   = (const float*)d_in[9];
  const float* ln2_w    = (const float*)d_in[10];
  const float* ffn_wg   = (const float*)d_in[11];
  const float* ffn_wu   = (const float*)d_in[12];
  const float* ffn_wd   = (const float*)d_in[13];
  const unsigned char* syn_mask = (const unsigned char*)d_in[14];
  float* out = (float*)d_out;

  // workspace carving (256B aligned)
  char* base = (char*)d_ws;
  size_t off = 0;
  auto carve = [&](size_t bytes) -> char* {
    char* p = base + off;
    off += (bytes + 255) & ~(size_t)255;
    return p;
  };
  const size_t MT = (size_t)Bb * Tt;             // 4096 rows
  u16* a_bf   = (u16*)carve(MT * Dd * 2);
  u16* Wi_bf  = (u16*)carve((size_t)Dd * Dd * 2);
  u16* Wr_bf  = (u16*)carve((size_t)Dd * Dd * 2);
  u16* syn_bf = (u16*)carve((size_t)Dd * Dd * 2);
  u16* gate_bf= (u16*)carve((size_t)Dd * Dd * 2);
  u16* wg_bf  = (u16*)carve((size_t)Hh * Dd * 2);
  u16* wu_bf  = (u16*)carve((size_t)Hh * Dd * 2);
  u16* wd_bf  = (u16*)carve((size_t)Dd * Hh * 2);
  float* U    = (float*)carve(MT * Dd * 4);
  u16* s_bf   = (u16*)carve(MT * Dd * 2);
  float* xmid = (float*)carve(MT * Dd * 4);
  u16* y_bf   = (u16*)carve(MT * Dd * 2);
  u16* t_bf   = (u16*)carve(MT * Hh * 2);
  u16* hA     = (u16*)carve((size_t)16 * Dd * 2);
  u16* hB     = (u16*)carve((size_t)16 * Dd * 2);
  int* bar    = (int*)carve(256);

  // 1) weight conversion to bf16 (stateless; repeated every call by design)
  k_f32_to_bf16<<<512, 256, 0, stream>>>(liq_Wi, Wi_bf, Dd * Dd);
  k_f32_to_bf16<<<512, 256, 0, stream>>>(liq_Wr, Wr_bf, Dd * Dd);
  k_f32_to_bf16<<<512, 256, 0, stream>>>(gate_w, gate_bf, Dd * Dd);
  k_masked_bf16<<<512, 256, 0, stream>>>(syn_w, syn_mask, syn_bf, Dd * Dd);
  k_f32_to_bf16<<<2048, 256, 0, stream>>>(ffn_wg, wg_bf, Hh * Dd);
  k_f32_to_bf16<<<2048, 256, 0, stream>>>(ffn_wu, wu_bf, Hh * Dd);
  k_f32_to_bf16<<<2048, 256, 0, stream>>>(ffn_wd, wd_bf, Dd * Hh);
  k_init<<<16, 256, 0, stream>>>(hA, hB, bar);

  // 2) a = rmsnorm(x, ln1_w) -> bf16
  k_rmsnorm<<<(int)MT, 256, 0, stream>>>(x, ln1_w, a_bf);

  // 3) U = a @ Wi^T  (f32)
  k_gemm_u<<<dim3(Dd / 128, (int)MT / 128), 256, 0, stream>>>(a_bf, Wi_bf, U);

  // 4) persistent liquid recurrence + fused PLIF -> spikes (bf16)
  k_liquid_plif<<<Dd / 16, 64, 0, stream>>>(Wr_bf, U, liq_b, liq_tau, plif_tau,
                                            hA, hB, s_bf, bar);

  // 5) xmid = x + (s@syn^T) * sigmoid(s@gate^T + gate_b)
  k_syn_gate<<<dim3(Dd / 64, (int)MT / 128), 256, 0, stream>>>(s_bf, syn_bf, gate_bf,
                                                               gate_b, x, xmid);

  // 6) y = rmsnorm(xmid, ln2_w) -> bf16
  k_rmsnorm<<<(int)MT, 256, 0, stream>>>(xmid, ln2_w, y_bf);

  // 7) t = silu(y@wg^T) * (y@wu^T) -> bf16 (halves intermediate HBM traffic)
  k_ffn_up<<<dim3(Hh / 64, (int)MT / 128), 256, 0, stream>>>(y_bf, wg_bf, wu_bf, t_bf);

  // 8) out = xmid + t @ wd^T
  k_ffn_down<<<dim3(Dd / 128, (int)MT / 128), 256, 0, stream>>>(t_bf, wd_bf, xmid, out);
}
